// LocalAttn_56530359550026
// MI455X (gfx1250) — matrix operand order repaired and checked by reading, not compiled, not run on hardware
//
#include <hip/hip_runtime.h>

// Sliding-window attention forward for MI455X (gfx1250), bf16 WMMA path.
// B=64, S=4096, D=64, WINDOW=256 (half-window 128, 32 bins).

namespace {
constexpr int Bv      = 64;
constexpr int Sv      = 4096;
constexpr int Dv      = 64;
constexpr int HW      = 128;           // half window
constexpr int NBINS   = Sv / HW;       // 32
constexpr int NKEY    = 3 * HW;        // 384 keys per bin
constexpr int KSTRIDE = 72;            // padded bf16 row stride for K tile
constexpr int VSTRIDE = 392;           // padded bf16 row stride for V^T tile
}

typedef __attribute__((ext_vector_type(16))) __bf16 v16bf;
typedef __attribute__((ext_vector_type(8)))  float  v8f;

static __device__ __forceinline__ v16bf lds_load16(const __bf16* p) {
  v16bf r;
#pragma unroll
  for (int i = 0; i < 16; ++i) r[i] = p[i];
  return r;
}

__global__ __launch_bounds__(256) void swa_fwd_bf16_wmma(
    const float* __restrict__ q, const float* __restrict__ k,
    const float* __restrict__ v, float* __restrict__ out) {
  __shared__ __bf16 Ks[NKEY * KSTRIDE];   // K tile, row-major [key][d], padded
  __shared__ __bf16 Vt[Dv * VSTRIDE];     // V tile transposed [d][key], padded
  __shared__ __bf16 Pst[8 * 16 * 32];     // per-wave P staging (16 rows x 32 keys)

  const int bin  = blockIdx.x;            // 0..31
  const int b    = blockIdx.y;            // 0..63
  const int tid  = threadIdx.x;
  const int w    = tid >> 5;              // wave id 0..7 -> query rows w*16..+16
  const int lane = tid & 31;
  const int l16  = lane & 15;
  const int hi   = lane >> 4;             // half-wave select

  const size_t base = (size_t)b * Sv * Dv;

  // ---- cooperative K/V tile load with half-window roll, f32 -> bf16 ----
  for (int idx = tid; idx < NKEY * Dv; idx += 256) {
    const int row  = idx >> 6;                                  // key 0..383
    const int col  = idx & 63;                                  // d   0..63
    const int binK = (bin + (row >> 7) + NBINS - 1) & (NBINS - 1);
    const size_t g = base + ((size_t)binK * HW + (row & (HW - 1))) * Dv + col;
    Ks[row * KSTRIDE + col] = (__bf16)k[g];
    Vt[col * VSTRIDE + row] = (__bf16)v[g];
  }
  __syncthreads();

  // ---- per-wave Q A-fragments (16 rows x 64 d, split into two K=32 chunks),
  //      softmax scale 1/sqrt(64) folded in ----
  const float* qrow = q + base + ((size_t)bin * HW + w * 16 + l16) * Dv;
  v16bf aQ[2];
#pragma unroll
  for (int kc = 0; kc < 2; ++kc) {
#pragma unroll
    for (int e = 0; e < 16; ++e) {
      const int dd = hi * 16 + e; // A layout
      aQ[kc][e] = (__bf16)(qrow[kc * 32 + dd] * 0.125f);
    }
  }

  const v8f vzero = {0.f, 0.f, 0.f, 0.f, 0.f, 0.f, 0.f, 0.f};
  v8f O[4];                                // output acc: 4 tiles of 16 d-cols
#pragma unroll
  for (int nt = 0; nt < 4; ++nt) O[nt] = vzero;
  float rs[8];                             // per-lane partial row sums
#pragma unroll
  for (int r = 0; r < 8; ++r) rs[r] = 0.f;

  // ---- stream over 12 super-tiles of 32 keys each ----
  for (int st = 0; st < 12; ++st) {
    // scores for two 16-key column tiles
    v8f S[2];
#pragma unroll
    for (int h = 0; h < 2; ++h) {
      const int ct = st * 2 + h;
      v8f acc = vzero;
#pragma unroll
      for (int kc = 0; kc < 2; ++kc) {
        // B frag (32x16): K = d-offset = hi*16+e, N = key = ct*16 + l16
        v16bf bk = lds_load16(&Ks[(ct * 16 + l16) * KSTRIDE + kc * 32 + hi * 16]);
        acc = __builtin_amdgcn_wmma_f32_16x16x32_bf16(
            false, aQ[kc], false, bk, (short)0, acc, false, false);
      }
      S[h] = acc;
    }

    // mask + exp (no max-sub: scores are O(1)); stage P as bf16 in A-layout src
#pragma unroll
    for (int h = 0; h < 2; ++h) {
#pragma unroll
      for (int r = 0; r < 8; ++r) {
        const int m = r + 8 * hi;                 // local query row 0..15
        const int i = w * 16 + m;                 // query index in bin 0..127
        const int j = st * 32 + h * 16 + l16;     // key index 0..383
        const bool masked = (j < HW && j < i) || (j >= 2 * HW && (j - 2 * HW) >= i);
        const float p = masked ? 0.f : __expf(S[h][r]);
        rs[r] += p;
        Pst[w * 512 + m * 32 + h * 16 + l16] = (__bf16)p;
      }
    }
    // wave-private LDS RAW: DS ops are in-order within a wave; stop compiler reordering
    asm volatile("" ::: "memory");

    // read P back as A fragment (16 rows x 32 keys)
    v16bf pa;
#pragma unroll
    for (int e = 0; e < 16; ++e) {
      const int kk = hi * 16 + e;
      pa[e] = Pst[w * 512 + l16 * 32 + kk];
    }

    // PV: accumulate 16x64 output, 4 column tiles of 16
#pragma unroll
    for (int nt = 0; nt < 4; ++nt) {
      // B frag (32x16): K = key-offset = hi*16+e, N = d = nt*16 + l16
      v16bf bv = lds_load16(&Vt[(nt * 16 + l16) * VSTRIDE + st * 32 + hi * 16]);
      O[nt] = __builtin_amdgcn_wmma_f32_16x16x32_bf16(
          false, pa, false, bv, (short)0, O[nt], false, false);
    }
  }

  // ---- row-sum reduction across the 16-lane halves (rows r+8*hi) ----
  float inv[8];
#pragma unroll
  for (int r = 0; r < 8; ++r) {
    float s = rs[r];
    s += __shfl_xor(s, 1, 32);
    s += __shfl_xor(s, 2, 32);
    s += __shfl_xor(s, 4, 32);
    s += __shfl_xor(s, 8, 32);
    inv[r] = 1.0f / s;
  }

  // ---- normalize + store ----
  float* orow = out + base + ((size_t)bin * HW + w * 16) * Dv;
#pragma unroll
  for (int nt = 0; nt < 4; ++nt) {
#pragma unroll
    for (int r = 0; r < 8; ++r) {
      const int m = r + 8 * hi;
      orow[m * Dv + nt * 16 + l16] = O[nt][r] * inv[r];
    }
  }
}

extern "C" void kernel_launch(void* const* d_in, const int* in_sizes, int n_in,
                              void* d_out, int out_size, void* d_ws, size_t ws_size,
                              hipStream_t stream) {
  (void)in_sizes; (void)n_in; (void)out_size; (void)d_ws; (void)ws_size;
  const float* q = (const float*)d_in[0];
  const float* k = (const float*)d_in[1];
  const float* v = (const float*)d_in[2];
  float* out = (float*)d_out;
  dim3 grid(NBINS, Bv);   // (32 bins, 64 batch*heads)
  dim3 block(256);        // 8 wave32 waves
  swa_fwd_bf16_wmma<<<grid, block, 0, stream>>>(q, k, v, out);
}
